// SingleHeadAttention_4088808866231
// MI455X (gfx1250) — matrix operand order repaired and checked by reading, not compiled, not run on hardware
//
#include <hip/hip_runtime.h>

// Single-head causal attention, B=4, S=2048, D=1024, fp32 in/out.
// Strategy: bf16 WMMA (v_wmma_f32_16x16x32_bf16), f32 accum.
//   K0: one-shot fp32->bf16 conversion of x and the 3 weight matrices.
//   K1: QKV projection GEMM; bf16 tiles staged into double-buffered LDS with
//       global_load_async_to_lds_b128 (ASYNCcnt) so fetch of tile k+1 overlaps
//       WMMA on tile k. V written transposed (d-major) for the PV B-operand.
//   K2: per-16-row-query-tile online softmax (2 passes over keys <= diagonal).
//   K3: out = P @ V over (q-tile, 64-wide d-chunk) waves with causal s-limit.

#define S_LEN 2048
#define D_DIM 1024
#define BATCH 4

typedef __attribute__((ext_vector_type(16))) __bf16 v16bf;
typedef __attribute__((ext_vector_type(8)))  float  v8f;

union Frag { v16bf v; uint4 u[2]; };

__device__ __forceinline__ uint4 ldg16(const __bf16* p) {
  return *reinterpret_cast<const uint4*>(p);
}

__device__ __forceinline__ v8f vzero() {
  v8f z;
#pragma unroll
  for (int i = 0; i < 8; ++i) z[i] = 0.0f;
  return z;
}

__device__ __forceinline__ v8f wmma_bf16(const Frag& a, const Frag& b, v8f c) {
  return __builtin_amdgcn_wmma_f32_16x16x32_bf16(false, a.v, false, b.v,
                                                 (short)0, c, false, false);
}

__device__ __forceinline__ unsigned pack2(__bf16 a, __bf16 b) {
  union { __bf16 h[2]; unsigned u; } x;
  x.h[0] = a; x.h[1] = b;
  return x.u;
}

// Async 16B global->LDS copy (CDNA5 GLOBAL_LOAD_ASYNC_TO_LDS_B128, GVS mode:
// mem = SADDR(sgpr64) + VADDR(i32 byte off); LDS dst = VGPR lds_addr).
__device__ __forceinline__ void async_copy16(unsigned lds_addr,
                                             const __bf16* base,
                                             unsigned byte_off) {
  asm volatile("global_load_async_to_lds_b128 %0, %1, %2"
               :
               : "v"(lds_addr), "v"(byte_off), "s"(base)
               : "memory");
}

// ---------------------------------------------------------------------------
// Kernel 0: convert x (B*S*D) and Wk|Wq|Wv (3*D*D) from fp32 to bf16.
// ---------------------------------------------------------------------------
__global__ __launch_bounds__(256) void convert_kernel(
    const float* __restrict__ x, const float* __restrict__ Wk,
    const float* __restrict__ Wq, const float* __restrict__ Wv,
    __bf16* __restrict__ Xb, __bf16* __restrict__ Wb)
{
  const size_t XN = (size_t)BATCH * S_LEN * D_DIM;   // 8M elements
  const size_t WN = (size_t)D_DIM * D_DIM;           // 1M elements
  const size_t i = ((size_t)blockIdx.x * 256 + threadIdx.x) * 4;
  const float* src;
  __bf16* dst;
  size_t off;
  if (i < XN) {
    src = x; dst = Xb; off = i;
  } else {
    const size_t j = i - XN;
    const size_t p = j / WN;
    const size_t r = j % WN;
    src = (p == 0) ? Wk : (p == 1) ? Wq : Wv;
    dst = Wb + p * WN;
    off = r;
  }
  const float4 f = *reinterpret_cast<const float4*>(src + off);
  union { __bf16 h[4]; uint2 u; } o;
  o.h[0] = (__bf16)f.x; o.h[1] = (__bf16)f.y;
  o.h[2] = (__bf16)f.z; o.h[3] = (__bf16)f.w;
  *reinterpret_cast<uint2*>(dst + off) = o.u;
}

// ---------------------------------------------------------------------------
// Kernel 1: Q = x Wq^T, K = x Wk^T, V^T (d-major) = (x Wv^T)^T, all bf16.
// Grid: (B*S/64, D/64), 128 threads (4 waves, 2x2 over a 64x64 tile).
// bf16 tiles staged via async-to-LDS, double buffered.
// ---------------------------------------------------------------------------
__global__ __launch_bounds__(128) void qkv_gemm_kernel(
    const __bf16* __restrict__ Xb, const __bf16* __restrict__ Wb,
    __bf16* __restrict__ Qb, __bf16* __restrict__ Kb, __bf16* __restrict__ Vt)
{
  constexpr int LDT   = 40;            // 32-elem rows + 8 pad (16B-aligned rows)
  constexpr int TILE1 = 64 * LDT;      // one 64x32 tile (elements)
  constexpr int TILE4 = 4 * TILE1;     // x tile + 3 weight tiles
  __shared__ __bf16 smem[2 * TILE4];   // double buffered (40 KB)

  const int tid  = threadIdx.x;
  const int lane = tid & 31;
  const int wid  = tid >> 5;
  const int lm   = lane & 15;
  const int hi   = lane >> 4;
  const int waveM = wid >> 1, waveN = wid & 1;
  const int row0 = blockIdx.x * 64;    // flattened (b, s) row block
  const int col0 = blockIdx.y * 64;    // output feature block

  const int srow  = tid >> 1;          // 0..63 staging row
  const int shalf = tid & 1;           // which 16-wide half of 32 d's

  const unsigned ldsBase = (unsigned)(size_t)(void*)smem;

  v8f acc[3][2][2];
#pragma unroll
  for (int p = 0; p < 3; ++p)
#pragma unroll
    for (int mt = 0; mt < 2; ++mt)
#pragma unroll
      for (int nt = 0; nt < 2; ++nt) acc[p][mt][nt] = vzero();

  // issue async copies for K-step kk into buffer `buf` (8 x 16B per thread)
  auto issue = [&](int buf, int kk) {
#pragma unroll
    for (int c = 0; c < 2; ++c) {
      const unsigned lds_x =
          ldsBase + (unsigned)(buf * TILE4 + srow * LDT + shalf * 16 + c * 8) * 2;
      const unsigned off_x =
          (unsigned)(((row0 + srow) * D_DIM + kk * 32 + shalf * 16 + c * 8) * 2);
      async_copy16(lds_x, Xb, off_x);
    }
#pragma unroll
    for (int p = 0; p < 3; ++p)
#pragma unroll
      for (int c = 0; c < 2; ++c) {
        const unsigned lds_w =
            ldsBase + (unsigned)(buf * TILE4 + (p + 1) * TILE1 +
                                 srow * LDT + shalf * 16 + c * 8) * 2;
        const unsigned off_w =
            (unsigned)((p * D_DIM * D_DIM + (col0 + srow) * D_DIM +
                        kk * 32 + shalf * 16 + c * 8) * 2);
        async_copy16(lds_w, Wb, off_w);
      }
  };

  issue(0, 0);
  for (int kk = 0; kk < D_DIM / 32; ++kk) {
    if (kk + 1 < D_DIM / 32) {
      issue((kk + 1) & 1, kk + 1);
      asm volatile("s_wait_asynccnt 0x8" ::: "memory");   // current buffer done
    } else {
      asm volatile("s_wait_asynccnt 0x0" ::: "memory");
    }
    __syncthreads();

    const __bf16* xs = smem + (kk & 1) * TILE4;
    // A fragments: x rows (M = s), per-lane K-chunks {0..7,16..23}/{8..15,24..31}
    Frag af[2];
#pragma unroll
    for (int mt = 0; mt < 2; ++mt) {
      const __bf16* ap = xs + (waveM * 32 + mt * 16 + lm) * LDT + hi * 8;
      af[mt].u[0] = *reinterpret_cast<const uint4*>(ap);
      af[mt].u[1] = *reinterpret_cast<const uint4*>(ap + 16);
    }
    // B fragments: weight rows (N = out feature), per-lane K-chunks {0..7,16..23}/{8..15,24..31}
#pragma unroll
    for (int p = 0; p < 3; ++p) {
      const __bf16* wl = xs + (p + 1) * TILE1;
#pragma unroll
      for (int nt = 0; nt < 2; ++nt) {
        Frag bf;
        const __bf16* bp = wl + (waveN * 32 + nt * 16 + lm) * LDT + hi * 8;
        bf.u[0] = *reinterpret_cast<const uint4*>(bp);
        bf.u[1] = *reinterpret_cast<const uint4*>(bp + 16);
#pragma unroll
        for (int mt = 0; mt < 2; ++mt)
          acc[p][mt][nt] = wmma_bf16(af[mt], bf, acc[p][mt][nt]);
      }
    }
    __syncthreads();
  }

  // ---- store K (p=0) and Q (p=1) row-major bf16 ----
#pragma unroll
  for (int p = 0; p < 2; ++p) {
    __bf16* dst = (p == 0) ? Kb : Qb;
#pragma unroll
    for (int mt = 0; mt < 2; ++mt)
#pragma unroll
      for (int nt = 0; nt < 2; ++nt)
#pragma unroll
        for (int r = 0; r < 8; ++r) {
          const int row = row0 + waveM * 32 + mt * 16 + r + hi * 8;
          const int col = col0 + waveN * 32 + nt * 16 + lm;
          dst[(size_t)row * D_DIM + col] = (__bf16)acc[p][mt][nt][r];
        }
  }

  // ---- V: transpose 64x64 tile through LDS, store d-major (Vt[b][d][s]) ----
  __bf16* vb = smem;                   // reuse staging LDS, stride 72
#pragma unroll
  for (int mt = 0; mt < 2; ++mt)
#pragma unroll
    for (int nt = 0; nt < 2; ++nt)
#pragma unroll
      for (int r = 0; r < 8; ++r) {
        const int sl = waveM * 32 + mt * 16 + r + hi * 8;
        const int jl = waveN * 32 + nt * 16 + lm;
        vb[sl * 72 + jl] = (__bf16)acc[2][mt][nt][r];
      }
  __syncthreads();
  {
    const int b  = row0 / S_LEN;
    const int s0 = row0 % S_LEN;
    const int j  = tid >> 1;
    const int sh = tid & 1;
    unsigned dw[16];
#pragma unroll
    for (int i = 0; i < 16; ++i)
      dw[i] = pack2(vb[(sh * 32 + 2 * i) * 72 + j],
                    vb[(sh * 32 + 2 * i + 1) * 72 + j]);
    __bf16* dst = Vt + ((size_t)b * D_DIM + col0 + j) * S_LEN + s0 + sh * 32;
    uint4 q0 = {dw[0],  dw[1],  dw[2],  dw[3]};
    uint4 q1 = {dw[4],  dw[5],  dw[6],  dw[7]};
    uint4 q2 = {dw[8],  dw[9],  dw[10], dw[11]};
    uint4 q3 = {dw[12], dw[13], dw[14], dw[15]};
    *reinterpret_cast<uint4*>(dst)      = q0;
    *reinterpret_cast<uint4*>(dst + 8)  = q1;
    *reinterpret_cast<uint4*>(dst + 16) = q2;
    *reinterpret_cast<uint4*>(dst + 24) = q3;
  }
}

// ---------------------------------------------------------------------------
// Kernel 2: P = softmax(causal(Q K^T / sqrt(D))) as bf16. One wave per
// 16-row query tile; Q fragments pinned in registers; two passes over keys.
// ---------------------------------------------------------------------------
__device__ __forceinline__ v8f score_tile(const __bf16* __restrict__ Kb,
                                          size_t kbase, const Frag (&qf)[32],
                                          int hi) {
  v8f acc = vzero();
#pragma unroll
  for (int kk = 0; kk < 32; ++kk) {
    Frag kf;
    const __bf16* p = Kb + kbase + kk * 32 + hi * 8;
    kf.u[0] = ldg16(p);
    kf.u[1] = ldg16(p + 16);
    acc = wmma_bf16(qf[kk], kf, acc);
  }
  return acc;
}

__global__ __launch_bounds__(128) void attn_softmax_kernel(
    const __bf16* __restrict__ Qb, const __bf16* __restrict__ Kb,
    __bf16* __restrict__ Pb)
{
  const int tid = threadIdx.x, lane = tid & 31, wid = tid >> 5;
  const int w  = blockIdx.x * 4 + wid;          // 0..511
  const int b  = w >> 7;
  const int qt = w & 127;
  const int lm = lane & 15, hi = lane >> 4;
  const float scale = 0.03125f;                 // 1/sqrt(1024)

  Frag qf[32];
  const size_t qrow = ((size_t)b * S_LEN + qt * 16 + lm) * D_DIM;
#pragma unroll
  for (int kk = 0; kk < 32; ++kk) {
    const __bf16* p = Qb + qrow + kk * 32 + hi * 8;
    qf[kk].u[0] = ldg16(p);
    qf[kk].u[1] = ldg16(p + 16);
  }

  float rm[8], rl[8];
#pragma unroll
  for (int r = 0; r < 8; ++r) { rm[r] = -3.0e30f; rl[r] = 0.0f; }

  // ---- pass 1: online row max / sum-exp over key tiles 0..qt ----
  for (int st = 0; st <= qt; ++st) {
    const size_t kbase = ((size_t)b * S_LEN + st * 16 + lm) * D_DIM;
    v8f acc = score_tile(Kb, kbase, qf, hi);
    const bool diag = (st == qt);
    const int n = st * 16 + lm;
#pragma unroll
    for (int r = 0; r < 8; ++r) {
      float v = acc[r] * scale;
      if (diag && n > qt * 16 + r + hi * 8) v = -1.0e30f;
      float t = v;                               // row max across 16 lanes
      t = fmaxf(t, __shfl_xor(t, 1, 32));
      t = fmaxf(t, __shfl_xor(t, 2, 32));
      t = fmaxf(t, __shfl_xor(t, 4, 32));
      t = fmaxf(t, __shfl_xor(t, 8, 32));
      const float nm = fmaxf(rm[r], t);
      float s = __expf(v - nm);
      s += __shfl_xor(s, 1, 32);
      s += __shfl_xor(s, 2, 32);
      s += __shfl_xor(s, 4, 32);
      s += __shfl_xor(s, 8, 32);
      rl[r] = rl[r] * __expf(rm[r] - nm) + s;
      rm[r] = nm;
    }
  }

  float ri[8];
#pragma unroll
  for (int r = 0; r < 8; ++r) ri[r] = 1.0f / rl[r];

  // ---- pass 2: recompute scores, write normalized P (bf16) ----
  for (int st = 0; st <= qt; ++st) {
    const size_t kbase = ((size_t)b * S_LEN + st * 16 + lm) * D_DIM;
    v8f acc = score_tile(Kb, kbase, qf, hi);
    const bool diag = (st == qt);
    const int n = st * 16 + lm;
#pragma unroll
    for (int r = 0; r < 8; ++r) {
      const float v = acc[r] * scale;
      const bool masked = diag && (n > qt * 16 + r + hi * 8);
      const float pv = masked ? 0.0f : __expf(v - rm[r]) * ri[r];
      Pb[((size_t)(b * S_LEN + qt * 16 + r + hi * 8)) * S_LEN + st * 16 + lm] =
          (__bf16)pv;
    }
  }
  // zero-pad the next 16 columns so the K=32 PV loop can over-read safely
  if ((qt & 1) == 0) {
#pragma unroll
    for (int r = 0; r < 8; ++r)
      Pb[((size_t)(b * S_LEN + qt * 16 + r + hi * 8)) * S_LEN +
         (qt + 1) * 16 + lm] = (__bf16)0.0f;
  }
}

// ---------------------------------------------------------------------------
// Kernel 3: out = P @ V. One wave per (q-tile, 64-wide d-chunk); Vt is
// d-major so B-operand reads are contiguous along s. Causal s-limit.
// ---------------------------------------------------------------------------
__global__ __launch_bounds__(128) void attn_out_kernel(
    const __bf16* __restrict__ Pb, const __bf16* __restrict__ Vt,
    float* __restrict__ out)
{
  const int tid = threadIdx.x, lane = tid & 31, wid = tid >> 5;
  const int w  = blockIdx.x * 4 + wid;          // 0..8191
  const int dc = w & 15;
  const int qt = (w >> 4) & 127;
  const int b  = w >> 11;
  const int lm = lane & 15, hi = lane >> 4;

  v8f acc[4];
#pragma unroll
  for (int nb = 0; nb < 4; ++nb) acc[nb] = vzero();

  const size_t prow = ((size_t)b * S_LEN + qt * 16 + lm) * S_LEN;
  const int slim = (qt + 1) * 16;
  for (int ss = 0; ss * 32 < slim; ++ss) {
    Frag pf;
    const __bf16* pp = Pb + prow + ss * 32 + hi * 8;
    pf.u[0] = ldg16(pp);
    pf.u[1] = ldg16(pp + 16);
#pragma unroll
    for (int nb = 0; nb < 4; ++nb) {
      Frag vf;
      const __bf16* vp = Vt +
          ((size_t)b * D_DIM + dc * 64 + nb * 16 + lm) * S_LEN + ss * 32 + hi * 8;
      vf.u[0] = ldg16(vp);
      vf.u[1] = ldg16(vp + 16);
      acc[nb] = wmma_bf16(pf, vf, acc[nb]);
    }
  }

#pragma unroll
  for (int nb = 0; nb < 4; ++nb)
#pragma unroll
    for (int r = 0; r < 8; ++r) {
      const int q = qt * 16 + r + hi * 8;
      const int d = dc * 64 + nb * 16 + lm;
      out[((size_t)b * S_LEN + q) * D_DIM + d] = acc[nb][r];
    }
}

// ---------------------------------------------------------------------------
extern "C" void kernel_launch(void* const* d_in, const int* in_sizes, int n_in,
                              void* d_out, int out_size, void* d_ws, size_t ws_size,
                              hipStream_t stream)
{
  (void)in_sizes; (void)n_in; (void)out_size; (void)ws_size;
  const float* x  = (const float*)d_in[0];
  const float* Wk = (const float*)d_in[1];
  const float* Wq = (const float*)d_in[2];
  const float* Wv = (const float*)d_in[3];
  float* out = (float*)d_out;

  char* ws = (char*)d_ws;
  const size_t MB = 1024 * 1024;
  __bf16* Qb = (__bf16*)(ws + 0 * MB);    // 16 MB: Q bf16 (B*S x D)
  __bf16* Kb = (__bf16*)(ws + 16 * MB);   // 16 MB: K bf16 (B*S x D)
  __bf16* Vt = (__bf16*)(ws + 32 * MB);   // 16 MB: V^T bf16 (B x D x S)
  __bf16* Pb = (__bf16*)(ws + 48 * MB);   // 32 MB: P bf16 (B x S x S)
  __bf16* Xb = (__bf16*)(ws + 80 * MB);   // 16 MB: x bf16
  __bf16* Wb = (__bf16*)(ws + 96 * MB);   //  6 MB: {Wk,Wq,Wv} bf16

  const int convN = (BATCH * S_LEN * D_DIM + 3 * D_DIM * D_DIM) / 4 / 256;
  convert_kernel<<<convN, 256, 0, stream>>>(x, Wk, Wq, Wv, Xb, Wb);
  qkv_gemm_kernel<<<dim3(BATCH * S_LEN / 64, D_DIM / 64), 128, 0, stream>>>(
      Xb, Wb, Qb, Kb, Vt);
  attn_softmax_kernel<<<BATCH * (S_LEN / 16) / 4, 128, 0, stream>>>(Qb, Kb, Pb);
  attn_out_kernel<<<BATCH * (S_LEN / 16) * (D_DIM / 64) / 4, 128, 0, stream>>>(
      Pb, Vt, out);
}
